// GraphDecoder_48911087567354
// MI455X (gfx1250) — matrix operand order repaired and checked by reading, not compiled, not run on hardware
//
#include <hip/hip_runtime.h>

// ---------------------------------------------------------------------------
// GraphDecoder for MI455X (gfx1250): all GEMMs via v_wmma_f32_16x16x32_bf16,
// weights pre-transposed+converted to bf16 [out][in] so both A and B WMMA
// fragments are contiguous 16B loads per lane (per CDNA5 ISA 7.12.2 layouts).
// ---------------------------------------------------------------------------

typedef __bf16 bf16_t;
typedef __attribute__((ext_vector_type(16))) __bf16 v16bf;
typedef __attribute__((ext_vector_type(8)))  float  v8f;

#define BB      64
#define NTOK    128
#define HD      512
#define NHEADS  8
#define NLAYER  4
#define PD      128
#define NFP1    9           // NF + 1
#define MTOK    (BB*NTOK)   // 8192
#define XCW     (PD + HD)   // 640

enum { EPI_F32 = 0, EPI_BF16 = 1, EPI_EXPAND = 2, EPI_QK = 3, EPI_VT = 4, EPI_SIG = 5 };

union FragBF {
  v16bf v;
  uint4 q[2];
};

// ---------------------------------------------------------------------------
// Generic batched GEMM: C = epilogue(alpha * (A[MxK] * Bt[NxK]^T + bias))
// A, Bt in bf16 row-major (K contiguous). Block tile 128x128, 8 waves (4Mx2N),
// wave tile 32x64 = 2x4 WMMA 16x16 accumulators. K must be a multiple of 32.
// Fragment layouts (wave32, ISA 7.12.2):
//   A 16x32 bf16: lane l holds row M=l%16; k chunks [kk+8*(l/16), +8) and
//                 [kk+16+8*(l/16), +8)  -> two 16B loads, 32B apart.
//   B 32x16 bf16: lane l holds col N=l%16; k chunks [kk+8*(l/16), +8) and
//                 [kk+16+8*(l/16), +8)  -> two 16B loads, 32B apart.
//   C/D f32 16x16: vgpr r -> row 8*(l/16)+r, col l%16.
// ---------------------------------------------------------------------------
__global__ __launch_bounds__(256) void k_gemm_bf16(
    const bf16_t* __restrict__ A, const bf16_t* __restrict__ Bt,
    const float* __restrict__ bias, void* __restrict__ outp,
    int M, int Nout, int Npad, int K,
    int lda, int ldb, int ldc,
    long sA, long sB, int bdiv, long sC1, long sC2,
    float alpha, int mode)
{
  const int z = blockIdx.z;
  A  += (long)z * sA;
  Bt += (long)z * sB;
  const long coff = (long)(z / bdiv) * sC1 + (long)(z % bdiv) * sC2;

  const int tid  = threadIdx.x;
  const int lane = tid & 31;
  const int wave = tid >> 5;
  const int lr   = lane & 15;
  const int half = lane >> 4;
  const int wm   = wave & 3;   // wave row group (M)
  const int wn   = wave >> 2;  // wave col group (N)
  const int bM   = blockIdx.y * 128;
  const int bN   = blockIdx.x * 128;

  v8f acc[2][4];
  #pragma unroll
  for (int mi = 0; mi < 2; ++mi)
    #pragma unroll
    for (int ni = 0; ni < 4; ++ni)
      #pragma unroll
      for (int e = 0; e < 8; ++e)
        acc[mi][ni][e] = 0.0f;

  const bf16_t* aptr[2];
  #pragma unroll
  for (int mi = 0; mi < 2; ++mi) {
    int r = bM + wm*32 + mi*16 + lr;
    if (r > M - 1) r = M - 1;                 // clamp; stores are guarded
    aptr[mi] = A + (long)r * lda + half * 8;
  }
  const bf16_t* bptr[4];
  #pragma unroll
  for (int ni = 0; ni < 4; ++ni) {
    int c = bN + wn*64 + ni*16 + lr;
    if (c > Npad - 1) c = Npad - 1;           // clamp; stores are guarded
    bptr[ni] = Bt + (long)c * ldb + half * 8;
  }

  for (int kk = 0; kk < K; kk += 32) {
    FragBF a[2], b[4];
    #pragma unroll
    for (int mi = 0; mi < 2; ++mi) {
      const uint4* p = (const uint4*)(aptr[mi] + kk);
      a[mi].q[0] = p[0];   // k = kk + 8*half .. +8
      a[mi].q[1] = p[2];   // k = kk + 16 + 8*half .. +8
    }
    #pragma unroll
    for (int ni = 0; ni < 4; ++ni) {
      const uint4* p = (const uint4*)(bptr[ni] + kk);
      b[ni].q[0] = p[0];   // k = kk + 8*half .. +8
      b[ni].q[1] = p[2];   // k = kk + 16 + 8*half .. +8
    }
    #pragma unroll
    for (int mi = 0; mi < 2; ++mi)
      #pragma unroll
      for (int ni = 0; ni < 4; ++ni)
        acc[mi][ni] = __builtin_amdgcn_wmma_f32_16x16x32_bf16(
            false, a[mi].v, false, b[ni].v, (short)0, acc[mi][ni], false, false);
  }

  #pragma unroll
  for (int mi = 0; mi < 2; ++mi) {
    #pragma unroll
    for (int ni = 0; ni < 4; ++ni) {
      #pragma unroll
      for (int r = 0; r < 8; ++r) {
        const int row = bM + wm*32 + mi*16 + half*8 + r;
        const int col = bN + wn*64 + ni*16 + lr;
        if (row >= M || col >= Nout) continue;
        float v = acc[mi][ni][r];
        if (bias) v += bias[col];
        v *= alpha;
        switch (mode) {
          case EPI_F32:
            ((float*)outp)[coff + (long)row * ldc + col] = v;
            break;
          case EPI_BF16:
            ((bf16_t*)outp)[coff + (long)row * ldc + col] = (bf16_t)v;
            break;
          case EPI_SIG:
            ((float*)outp)[coff + (long)row * ldc + col] = 1.0f / (1.0f + __expf(-v));
            break;
          case EPI_EXPAND: {   // row = batch, col = n*512+h -> xc[(b*128+n)][128+h]
            float s = v / (1.0f + __expf(-v));
            long idx = ((long)row * NTOK + (col >> 9)) * XCW + PD + (col & (HD - 1));
            ((bf16_t*)outp)[idx] = (bf16_t)s;
          } break;
          case EPI_QK: {       // row = b*128+tok, col = h*512+e -> [b][h][tok][e]
            long bi = row >> 7, tok = row & 127, hh = col >> 9, e = col & 511;
            ((bf16_t*)outp)[(((bi * NHEADS + hh) * NTOK + tok) * HD) + e] = (bf16_t)v;
          } break;
          case EPI_VT: {       // transposed V: [b][h][e][tok]
            long bi = row >> 7, tok = row & 127, hh = col >> 9, e = col & 511;
            ((bf16_t*)outp)[(((bi * NHEADS + hh) * HD + e) * NTOK) + tok] = (bf16_t)v;
          } break;
        }
      }
    }
  }
}

// fp32 [K][Nin] -> bf16 [Nout][K] (Nout>=Nin rows zero-filled), batched.
__global__ void k_transpose_conv(const float* __restrict__ in, bf16_t* __restrict__ out,
                                 int K, int Nin, int Nout, long sIn, long sOut)
{
  __shared__ float tile[32][33];
  in  += (long)blockIdx.z * sIn;
  out += (long)blockIdx.z * sOut;
  const int n0 = blockIdx.x * 32;
  const int k0 = blockIdx.y * 32;
  #pragma unroll
  for (int i = 0; i < 4; ++i) {
    int k = k0 + threadIdx.y + i * 8;
    int n = n0 + threadIdx.x;
    float v = (k < K && n < Nin) ? in[(long)k * Nin + n] : 0.0f;
    tile[threadIdx.y + i * 8][threadIdx.x] = v;
  }
  __syncthreads();
  #pragma unroll
  for (int i = 0; i < 4; ++i) {
    int n = n0 + threadIdx.y + i * 8;
    int k = k0 + threadIdx.x;
    if (n < Nout && k < K)
      out[(long)n * K + k] = (bf16_t)tile[threadIdx.x][threadIdx.y + i * 8];
  }
}

__global__ void k_f32_to_bf16(const float* __restrict__ in, bf16_t* __restrict__ out, long n)
{
  long i = (long)blockIdx.x * blockDim.x + threadIdx.x;
  if (i < n) out[i] = (bf16_t)in[i];
}

// y = post( r1? + r2? + LN_512(in)*g+be ), post = silu if do_silu. One row/block.
__global__ __launch_bounds__(256) void k_ln512(
    const float* __restrict__ in, const float* __restrict__ g, const float* __restrict__ be,
    const float* __restrict__ r1, const float* __restrict__ r2, int do_silu,
    float* __restrict__ out_f, bf16_t* __restrict__ out_b)
{
  __shared__ float s1[256], s2[256];
  const long base = (long)blockIdx.x * HD;
  const int t = threadIdx.x;
  float sa = 0.f, sb = 0.f;
  #pragma unroll
  for (int j = t; j < HD; j += 256) { float v = in[base + j]; sa += v; sb += v * v; }
  s1[t] = sa; s2[t] = sb;
  __syncthreads();
  for (int off = 128; off > 0; off >>= 1) {
    if (t < off) { s1[t] += s1[t + off]; s2[t] += s2[t + off]; }
    __syncthreads();
  }
  const float mean = s1[0] * (1.0f / HD);
  const float var  = s2[0] * (1.0f / HD) - mean * mean;
  const float rinv = rsqrtf(var + 1e-3f);
  #pragma unroll
  for (int j = t; j < HD; j += 256) {
    float v = in[base + j];
    float y = g[j] * (v - mean) * rinv + be[j];
    if (r1) y += r1[base + j];
    if (r2) y += r2[base + j];
    if (do_silu) y = y / (1.0f + __expf(-y));
    if (out_f) out_f[base + j] = y;
    if (out_b) out_b[base + j] = (bf16_t)y;
  }
}

// softmax over rows of 128 (already scaled), fp32 in -> bf16 out.
__global__ __launch_bounds__(128) void k_softmax128(const float* __restrict__ in,
                                                    bf16_t* __restrict__ out)
{
  __shared__ float sh[128];
  const long base = (long)blockIdx.x * 128;
  const int j = threadIdx.x;
  float v = in[base + j];
  sh[j] = v; __syncthreads();
  for (int off = 64; off > 0; off >>= 1) { if (j < off) sh[j] = fmaxf(sh[j], sh[j + off]); __syncthreads(); }
  float mx = sh[0];
  __syncthreads();
  float e = __expf(v - mx);
  sh[j] = e; __syncthreads();
  for (int off = 64; off > 0; off >>= 1) { if (j < off) sh[j] += sh[j + off]; __syncthreads(); }
  out[base + j] = (bf16_t)(e / sh[0]);
}

// softmax over 9 (input rows padded to 16 floats)
__global__ void k_softmax9(const float* __restrict__ in, float* __restrict__ out)
{
  int m = blockIdx.x * blockDim.x + threadIdx.x;
  if (m >= MTOK) return;
  const float* r = in + (long)m * 16;
  float mx = r[0];
  for (int j = 1; j < NFP1; ++j) mx = fmaxf(mx, r[j]);
  float e[NFP1], s = 0.f;
  for (int j = 0; j < NFP1; ++j) { e[j] = __expf(r[j] - mx); s += e[j]; }
  float inv = 1.0f / s;
  for (int j = 0; j < NFP1; ++j) out[(long)m * NFP1 + j] = e[j] * inv;
}

__device__ __forceinline__ float ln128_helper(float v, int j, const float* g, const float* be,
                                              float* s1, float* s2)
{
  s1[j] = v; s2[j] = v * v;
  __syncthreads();
  for (int off = 64; off > 0; off >>= 1) {
    if (j < off) { s1[j] += s1[j + off]; s2[j] += s2[j + off]; }
    __syncthreads();
  }
  float mean = s1[0] * (1.0f / PD);
  float var  = s2[0] * (1.0f / PD) - mean * mean;
  float r = rsqrtf(var + 1e-3f);
  __syncthreads();
  return g[j] * (v - mean) * r + be[j];
}

__global__ __launch_bounds__(128) void k_pos_init(const float* __restrict__ W0,
                                                  const float* __restrict__ b0,
                                                  const float* __restrict__ g,
                                                  const float* __restrict__ be,
                                                  float* __restrict__ out)
{
  __shared__ float s1[128], s2[128];
  int n = blockIdx.x, j = threadIdx.x;
  float p = (float)n * (1.0f / NTOK);
  float v = p * W0[j] + b0[j];
  out[(long)n * PD + j] = ln128_helper(v, j, g, be, s1, s2);
}

__global__ __launch_bounds__(128) void k_pos_mm(const float* __restrict__ in,
                                                const float* __restrict__ W,
                                                const float* __restrict__ b,
                                                const float* __restrict__ g,
                                                const float* __restrict__ be,
                                                float* __restrict__ out)
{
  __shared__ float rowv[128], s1[128], s2[128];
  int n = blockIdx.x, j = threadIdx.x;
  rowv[j] = in[(long)n * PD + j];
  __syncthreads();
  float accv = b[j];
  for (int k2 = 0; k2 < PD; ++k2) accv += rowv[k2] * W[k2 * PD + j];
  out[(long)n * PD + j] = ln128_helper(accv, j, g, be, s1, s2);
}

__global__ __launch_bounds__(128) void k_pos_fill(const float* __restrict__ pos,
                                                  bf16_t* __restrict__ xc)
{
  int m = blockIdx.x, j = threadIdx.x;
  xc[(long)m * XCW + j] = (bf16_t)pos[(long)(m & (NTOK - 1)) * PD + j];
}

// ---------------------------------------------------------------------------
extern "C" void kernel_launch(void* const* d_in, const int* in_sizes, int n_in,
                              void* d_out, int out_size, void* d_ws, size_t ws_size,
                              hipStream_t stream)
{
  (void)in_sizes; (void)n_in; (void)out_size; (void)ws_size;
  auto F = [&](int i) { return (const float*)d_in[i]; };
  const float* Z        = F(0);
  const float* W_expand = F(1);  const float* b_expand = F(2);
  const float* W_pos0   = F(3);  const float* b_pos0   = F(4);
  const float* W_pos12  = F(5);  const float* b_pos12  = F(6);
  const float* g_pos    = F(7);  const float* be_pos   = F(8);
  const float* W_comb   = F(9);  const float* b_comb   = F(10);
  const float* g_comb   = F(11); const float* be_comb  = F(12);
  const float* Wq = F(13); const float* bq = F(14);
  const float* Wk = F(15); const float* bk = F(16);
  const float* Wv = F(17); const float* bv = F(18);
  const float* Wo1 = F(19); const float* bo1 = F(20);
  const float* g1  = F(21); const float* be1 = F(22);
  const float* Wo2 = F(23); const float* bo2 = F(24);
  const float* g2  = F(25); const float* be2 = F(26);
  const float* Wo3 = F(27); const float* bo3 = F(28);
  const float* g3  = F(29); const float* be3 = F(30);
  const float* W_adj = F(31); const float* b_adj = F(32);
  const float* W_nf  = F(33); const float* b_nf  = F(34);

  // ---- workspace carve-up (deterministic every call) ----
  char* wp = (char*)d_ws;
  auto alloc = [&](size_t bytes) -> void* {
    void* p = (void*)wp;
    wp += (bytes + 255) & ~(size_t)255;
    return p;
  };
  bf16_t* wexp_t  = (bf16_t*)alloc((size_t)65536 * 512 * 2);
  bf16_t* wcomb_t = (bf16_t*)alloc((size_t)512 * XCW * 2);
  bf16_t* wq_t    = (bf16_t*)alloc((size_t)NLAYER * NHEADS * 512 * 512 * 2);
  bf16_t* wk_t    = (bf16_t*)alloc((size_t)NLAYER * NHEADS * 512 * 512 * 2);
  bf16_t* wv_t    = (bf16_t*)alloc((size_t)NLAYER * NHEADS * 512 * 512 * 2);
  bf16_t* wo1_t   = (bf16_t*)alloc((size_t)NLAYER * 512 * 4096 * 2);
  bf16_t* wo2_t   = (bf16_t*)alloc((size_t)NLAYER * 512 * 512 * 2);
  bf16_t* wo3_t   = (bf16_t*)alloc((size_t)NLAYER * 512 * 512 * 2);
  bf16_t* wadj_t  = (bf16_t*)alloc((size_t)128 * 512 * 2);
  bf16_t* wnf_t   = (bf16_t*)alloc((size_t)16 * 512 * 2);
  bf16_t* z_bf    = (bf16_t*)alloc((size_t)BB * 512 * 2);
  bf16_t* xc_bf   = (bf16_t*)alloc((size_t)MTOK * XCW * 2);
  float*  tmp512  = (float*) alloc((size_t)MTOK * HD * 4);
  float*  x_f     = (float*) alloc((size_t)MTOK * HD * 4);
  bf16_t* x_bf    = (bf16_t*)alloc((size_t)MTOK * HD * 2);
  float*  a_f     = (float*) alloc((size_t)MTOK * HD * 4);
  bf16_t* a_bf    = (bf16_t*)alloc((size_t)MTOK * HD * 2);
  float*  b2_f    = (float*) alloc((size_t)MTOK * HD * 4);
  bf16_t* b2_bf   = (bf16_t*)alloc((size_t)MTOK * HD * 2);
  bf16_t* q_bf    = (bf16_t*)alloc((size_t)BB * NHEADS * NTOK * HD * 2);
  bf16_t* k_bf    = (bf16_t*)alloc((size_t)BB * NHEADS * NTOK * HD * 2);
  bf16_t* vt_bf   = (bf16_t*)alloc((size_t)BB * NHEADS * HD * NTOK * 2);
  float*  sc_f    = (float*) alloc((size_t)BB * NHEADS * NTOK * NTOK * 4);
  bf16_t* sc_bf   = (bf16_t*)alloc((size_t)BB * NHEADS * NTOK * NTOK * 2);
  float*  pos_a   = (float*) alloc((size_t)NTOK * PD * 4);
  float*  pos_b   = (float*) alloc((size_t)NTOK * PD * 4);
  bf16_t* ctx_bf  = q_bf;   // alias: q dead once scores computed
  float*  nf_raw  = sc_f;   // alias: scores dead after the layer loop

  float* adj_out = (float*)d_out;                       // [8192][128]
  float* nf_out  = adj_out + (size_t)MTOK * NTOK;       // [8192][9]
  const float scale = 0.044194173824159216f;            // 1/sqrt(512)

  auto gemm = [&](const bf16_t* A, const bf16_t* Bt, const float* bias, void* out,
                  int M, int Nout, int Npad, int K, int lda, int ldb, int ldc,
                  float alpha, int mode, int batch = 1, long sA = 0, long sB = 0,
                  int bdiv = 1, long sC1 = 0, long sC2 = 0) {
    dim3 grid((Npad + 127) / 128, (M + 127) / 128, batch);
    k_gemm_bf16<<<grid, dim3(256), 0, stream>>>(A, Bt, bias, out, M, Nout, Npad, K,
                                                lda, ldb, ldc, sA, sB, bdiv, sC1, sC2,
                                                alpha, mode);
  };
  auto tconv = [&](const float* in, bf16_t* out, int K, int Nin, int Nout,
                   int batch, long sIn, long sOut) {
    dim3 grid((Nout + 31) / 32, (K + 31) / 32, batch);
    k_transpose_conv<<<grid, dim3(32, 8), 0, stream>>>(in, out, K, Nin, Nout, sIn, sOut);
  };
  auto ln = [&](const float* in, const float* g, const float* be, const float* r1,
                const float* r2, int silu, float* of, bf16_t* ob) {
    k_ln512<<<dim3(MTOK), dim3(256), 0, stream>>>(in, g, be, r1, r2, silu, of, ob);
  };

  // ---- weight prep: fp32 [K][N] -> bf16 [N][K] ----
  tconv(W_expand, wexp_t, 512, 65536, 65536, 1, 0, 0);
  tconv(W_comb,  wcomb_t, XCW, 512, 512, 1, 0, 0);
  tconv(Wq, wq_t, 512, 512, 512, NLAYER * NHEADS, 512L * 512, 512L * 512);
  tconv(Wk, wk_t, 512, 512, 512, NLAYER * NHEADS, 512L * 512, 512L * 512);
  tconv(Wv, wv_t, 512, 512, 512, NLAYER * NHEADS, 512L * 512, 512L * 512);
  tconv(Wo1, wo1_t, 4096, 512, 512, NLAYER, 4096L * 512, 512L * 4096);
  tconv(Wo2, wo2_t, 512, 512, 512, NLAYER, 512L * 512, 512L * 512);
  tconv(Wo3, wo3_t, 512, 512, 512, NLAYER, 512L * 512, 512L * 512);
  tconv(W_adj, wadj_t, 512, 128, 128, 1, 0, 0);
  tconv(W_nf, wnf_t, 512, NFP1, 16, 1, 0, 0);   // zero-padded to 16 cols
  k_f32_to_bf16<<<dim3((BB * 512 + 255) / 256), dim3(256), 0, stream>>>(Z, z_bf, (long)BB * 512);

  // ---- positional MLP (tiny, VALU) + fill of concat buffer ----
  k_pos_init<<<dim3(NTOK), dim3(PD), 0, stream>>>(W_pos0, b_pos0, g_pos, be_pos, pos_a);
  k_pos_mm<<<dim3(NTOK), dim3(PD), 0, stream>>>(pos_a, W_pos12,            b_pos12,
                                                g_pos + PD,     be_pos + PD,     pos_b);
  k_pos_mm<<<dim3(NTOK), dim3(PD), 0, stream>>>(pos_b, W_pos12 + PD * PD,  b_pos12 + PD,
                                                g_pos + 2 * PD, be_pos + 2 * PD, pos_a);
  k_pos_fill<<<dim3(MTOK), dim3(PD), 0, stream>>>(pos_a, xc_bf);

  // ---- expand: silu(Z @ W_expand + b) scattered into xc[:,128:640] ----
  gemm(z_bf, wexp_t, b_expand, xc_bf, BB, 65536, 65536, 512, 512, 512, 0, 1.0f, EPI_EXPAND);

  // ---- comb: LN(xc @ W_comb + b) -> x ----
  gemm(xc_bf, wcomb_t, b_comb, tmp512, MTOK, 512, 512, XCW, XCW, XCW, 512, 1.0f, EPI_F32);
  ln(tmp512, g_comb, be_comb, nullptr, nullptr, 0, x_f, x_bf);

  // ---- 4 attention layers ----
  for (int l = 0; l < NLAYER; ++l) {
    const long wqkv_off = (long)l * NHEADS * 512 * 512;
    // q/k/v projections (per-head full-width), fused bias + layout scatter
    gemm(x_bf, wq_t + wqkv_off, bq + l * 4096, q_bf,
         MTOK, 4096, 4096, 512, 512, 512, 0, 1.0f, EPI_QK);
    gemm(x_bf, wk_t + wqkv_off, bk + l * 4096, k_bf,
         MTOK, 4096, 4096, 512, 512, 512, 0, 1.0f, EPI_QK);
    gemm(x_bf, wv_t + wqkv_off, bv + l * 4096, vt_bf,
         MTOK, 4096, 4096, 512, 512, 512, 0, 1.0f, EPI_VT);
    // scores = scale * q k^T   (512 batches of 128x128x512)
    gemm(q_bf, k_bf, nullptr, sc_f, NTOK, NTOK, NTOK, HD, HD, HD, NTOK,
         scale, EPI_F32, BB * NHEADS, (long)NTOK * HD, (long)NTOK * HD,
         1, (long)NTOK * NTOK, 0);
    k_softmax128<<<dim3(BB * NHEADS * NTOK), dim3(128), 0, stream>>>(sc_f, sc_bf);
    // ctx = softmax(scores) @ v   -> concat layout [m][h*512+e] (bf16)
    gemm(sc_bf, vt_bf, nullptr, ctx_bf, NTOK, HD, HD, NTOK, NTOK, NTOK, NHEADS * HD,
         1.0f, EPI_BF16, BB * NHEADS, (long)NTOK * NTOK, (long)HD * NTOK,
         NHEADS, (long)NTOK * NHEADS * HD, (long)HD);
    // a = silu(x + LN(ctx @ Wo1 + bo1))
    gemm(ctx_bf, wo1_t + (long)l * 512 * 4096, bo1 + l * 512, tmp512,
         MTOK, 512, 512, 4096, 4096, 4096, 512, 1.0f, EPI_F32);
    ln(tmp512, g1 + l * 512, be1 + l * 512, x_f, nullptr, 1, a_f, a_bf);
    // b2 = silu(a + LN(a @ Wo2 + bo2))
    gemm(a_bf, wo2_t + (long)l * 512 * 512, bo2 + l * 512, tmp512,
         MTOK, 512, 512, 512, 512, 512, 512, 1.0f, EPI_F32);
    ln(tmp512, g2 + l * 512, be2 + l * 512, a_f, nullptr, 1, b2_f, b2_bf);
    // x = silu(a + b2 + LN(b2 @ Wo3 + bo3))
    gemm(b2_bf, wo3_t + (long)l * 512 * 512, bo3 + l * 512, tmp512,
         MTOK, 512, 512, 512, 512, 512, 512, 1.0f, EPI_F32);
    ln(tmp512, g3 + l * 512, be3 + l * 512, a_f, b2_f, 1, x_f, x_bf);
  }

  // ---- heads ----
  gemm(x_bf, wadj_t, b_adj, adj_out, MTOK, 128, 128, 512, 512, 512, 128,
       scale, EPI_SIG);
  gemm(x_bf, wnf_t, b_nf, nf_raw, MTOK, NFP1, 16, 512, 512, 512, 16,
       scale, EPI_F32);
  k_softmax9<<<dim3((MTOK + 255) / 256), dim3(256), 0, stream>>>(nf_raw, nf_out);
}
